// DNNNeuron_35777077575959
// MI455X (gfx1250) — matrix operand order repaired and checked by reading, not compiled, not run on hardware
//
#include <hip/hip_runtime.h>

typedef __attribute__((ext_vector_type(16))) _Float16 v16h;
typedef __attribute__((ext_vector_type(8)))  _Float16 v8h;
typedef __attribute__((ext_vector_type(2)))  __fp16   fp16x2;   // cvt_pkrtz result type
typedef __attribute__((ext_vector_type(8)))  float    v8f;
typedef __attribute__((ext_vector_type(4)))  float    f4;
typedef __attribute__((ext_vector_type(2)))  float    f2;

#define HDIM   10
#define NMID   4
#define EPSLN  1e-5f
#define LEAK   0.01f
#define NWAVES 8      // 256-thread block = 8 wave32
#define ROWS   32     // rows per wave per iteration (two 16-row WMMA tiles)
#define LDSW   4096   // per-wave: 2KB f16 A-tile [32][32] + 2KB f32 D-tile [32][16]

// single-instruction ReLU: v_med3_num_f32(x, 0, +inf)
__device__ __forceinline__ float relu1(float x) {
  return __builtin_amdgcn_fmed3f(x, 0.f, __builtin_inff());
}

__global__ __launch_bounds__(256) void mlp10_wmma(
    const float* __restrict__ hidden,
    const float* __restrict__ W_in,  const float* __restrict__ b_in,
    const float* __restrict__ ln_g,  const float* __restrict__ ln_b,
    const float* __restrict__ W_mid, const float* __restrict__ b_mid,
    const float* __restrict__ W_out, const float* __restrict__ b_out,
    float* __restrict__ out, long long N)
{
  __shared__ __align__(16) unsigned char smem[NWAVES * LDSW];
  const int lane = threadIdx.x & 31;
  const int wave = threadIdx.x >> 5;
  _Float16* At = (_Float16*)(smem + wave * LDSW);          // [32 rows][32 halves]
  float*    Dt = (float*)   (smem + wave * LDSW + 2048);   // [32 rows][16 f32]
  const int n  = lane & 15;   // output-column / N index
  const int hi = lane >> 4;   // wave half

  // ---- zero A-tile once: halves K=10..31 must stay zero forever (B pad rows are
  //      zero, but WMMA requires finite A pads: NaN*0 = NaN). Full-width rows let
  //      the WMMA A operand (incl. zero pad) be loaded straight from LDS.
  {
    f4 z = {0.f, 0.f, 0.f, 0.f};
    *(f4*)(At + lane * 32)      = z;
    *(f4*)(At + lane * 32 + 8)  = z;
    *(f4*)(At + lane * 32 + 16) = z;
    *(f4*)(At + lane * 32 + 24) = z;
  }

  // ---- hoist small uniform params into registers
  float win[HDIM], bin[HDIM], wo[HDIM];
  #pragma unroll
  for (int j = 0; j < HDIM; ++j) { win[j] = W_in[j]; bin[j] = b_in[j]; wo[j] = W_out[j]; }
  const float bo = b_out[0];

  // ---- fold LayerNorm affine into the GEMM:
  //      relu(LN(x) @ W + b) = relu( xhat @ (diag(g)W) + (beta@W + b) )
  // B fragment layout for v_wmma_f32_16x16x32_f16 (32x16 B):
  //   lanes 0-15 hold K=0..7 (e<8) and K=16..23 (e>=8); lanes 16-31 hold K=8..15 and K=24..31 (K>=10 pad -> 0).
  v16h Bf[NMID];
  float Cb[NMID];
  #pragma unroll
  for (int i = 0; i < NMID; ++i) {
    v16h b;
    #pragma unroll
    for (int e = 0; e < 16; ++e) b[e] = (_Float16)0.f;
    float c = 0.f;
    if (n < HDIM) {
      #pragma unroll
      for (int k = 0; k < HDIM; ++k) {
        float w = W_mid[i * HDIM * HDIM + k * HDIM + n];
        c = fmaf(ln_b[i * HDIM + k], w, c);
        if (hi == (k >> 3)) b[k & 7] = (_Float16)(ln_g[i * HDIM + k] * w);
      }
      c += b_mid[i * HDIM + n];
    }
    Bf[i] = b;
    Cb[i] = c;
  }

  const long long stride = (long long)gridDim.x * NWAVES * ROWS;
  for (long long rb = ((long long)blockIdx.x * NWAVES + wave) * ROWS;
       rb + ROWS <= N; rb += stride)   // N is a multiple of 32 (2^23)
  {
    const float h = hidden[rb + lane];         // row-per-lane: lane L owns row rb+L
    float x[HDIM];
    #pragma unroll
    for (int j = 0; j < HDIM; ++j) x[j] = fmaf(h, win[j], bin[j]);

    #pragma unroll
    for (int i = 0; i < NMID; ++i) {
      // ---- LayerNorm stats, fully per-lane
      float s = x[0];
      #pragma unroll
      for (int j = 1; j < HDIM; ++j) s += x[j];
      const float mu = s * (1.f / HDIM);
      float v = 0.f;
      #pragma unroll
      for (int j = 0; j < HDIM; ++j) { float d = x[j] - mu; v = fmaf(d, d, v); }
      const float rs = rsqrtf(v * (1.f / HDIM) + EPSLN);

      // ---- pack normalized row to f16 A-tile (row = lane, halves 0..9; 10..31 stay 0)
      fp16x2* arow = (fp16x2*)(At + lane * 32);
      #pragma unroll
      for (int p = 0; p < HDIM / 2; ++p)
        arow[p] = __builtin_amdgcn_cvt_pkrtz((x[2*p] - mu) * rs, (x[2*p+1] - mu) * rs);
      __builtin_amdgcn_wave_barrier();
      asm volatile("s_wait_dscnt 0" ::: "memory");

      // ---- A fragments straight from LDS (incl. zero pad region):
      //      elems 0..7 : lane<16 -> K0..7  | lane>=16 -> K8..15   (bytes  hi*16)
      //      elems 8..15: lane<16 -> K16..23| lane>=16 -> K24..31  (bytes 32+hi*16)
      v8h a0lo = *(const v8h*)(At + n * 32        + hi * 8);
      v8h a0hi = *(const v8h*)(At + n * 32 + 16   + hi * 8);
      v8h a1lo = *(const v8h*)(At + (16 + n) * 32      + hi * 8);
      v8h a1hi = *(const v8h*)(At + (16 + n) * 32 + 16 + hi * 8);
      v16h a0 = __builtin_shufflevector(a0lo, a0hi, 0,1,2,3,4,5,6,7,8,9,10,11,12,13,14,15);
      v16h a1 = __builtin_shufflevector(a1lo, a1hi, 0,1,2,3,4,5,6,7,8,9,10,11,12,13,14,15);

      // ---- accumulators pre-loaded with folded bias (C layout: per-lane column n)
      v8f c0, c1;
      #pragma unroll
      for (int r = 0; r < 8; ++r) { c0[r] = Cb[i]; c1[r] = Cb[i]; }

      v8f d0 = __builtin_amdgcn_wmma_f32_16x16x32_f16(false, a0, false, Bf[i],
                                                      (short)0, c0, false, false);
      v8f d1 = __builtin_amdgcn_wmma_f32_16x16x32_f16(false, a1, false, Bf[i],
                                                      (short)0, c1, false, false);

      // ---- ReLU (v_med3) + scatter C-layout D (vgpr r -> M=r+8*hi, N=n) to f32 tile
      #pragma unroll
      for (int r = 0; r < 8; ++r) {
        Dt[(hi * 8 + r) * 16 + n]      = relu1(d0[r]);
        Dt[(16 + hi * 8 + r) * 16 + n] = relu1(d1[r]);
      }
      __builtin_amdgcn_wave_barrier();
      asm volatile("s_wait_dscnt 0" ::: "memory");

      // ---- gather back row-per-lane
      const float* drow = Dt + lane * 16;
      f4 q0 = *(const f4*)drow;
      f4 q1 = *(const f4*)(drow + 4);
      f2 q2 = *(const f2*)(drow + 8);
      #pragma unroll
      for (int j = 0; j < 4; ++j) { x[j] = q0[j]; x[4 + j] = q1[j]; }
      x[8] = q2[0]; x[9] = q2[1];
    }

    // ---- output head: dot + bias + residual, LeakyTanh
    float acc = bo;
    #pragma unroll
    for (int j = 0; j < HDIM; ++j) acc = fmaf(x[j], wo[j], acc);
    acc += h;
    out[rb + lane] = tanhf(acc) + LEAK * acc;
  }
}

extern "C" void kernel_launch(void* const* d_in, const int* in_sizes, int n_in,
                              void* d_out, int out_size, void* d_ws, size_t ws_size,
                              hipStream_t stream)
{
  const float* hidden = (const float*)d_in[0];
  const float* W_in   = (const float*)d_in[1];
  const float* b_in   = (const float*)d_in[2];
  const float* ln_g   = (const float*)d_in[3];
  const float* ln_b   = (const float*)d_in[4];
  const float* W_mid  = (const float*)d_in[5];
  const float* b_mid  = (const float*)d_in[6];
  const float* W_out  = (const float*)d_in[7];
  const float* b_out  = (const float*)d_in[8];
  const long long N   = (long long)in_sizes[0];

  long long chunks = (N + (long long)NWAVES * ROWS - 1) / ((long long)NWAVES * ROWS);
  int grid = 4096;                       // 32768 waves; 8 grid-stride iters at N=2^23
  if (chunks < grid) grid = (int)(chunks > 0 ? chunks : 1);

  mlp10_wmma<<<grid, 256, 0, stream>>>(hidden, W_in, b_in, ln_g, ln_b,
                                       W_mid, b_mid, W_out, b_out,
                                       (float*)d_out, N);
}
